// Block_2697239462764
// MI455X (gfx1250) — matrix operand order repaired and checked by reading, not compiled, not run on hardware
//
#include <hip/hip_runtime.h>
#include <math.h>

// ---------------------------------------------------------------------------
// Types for CDNA5 WMMA (wave32, 16x16x32 bf16 -> f32)
// ---------------------------------------------------------------------------
typedef __bf16 bf16;
typedef __attribute__((ext_vector_type(8)))  __bf16 v8bf;
typedef __attribute__((ext_vector_type(16))) __bf16 v16bf;
typedef __attribute__((ext_vector_type(8)))  float  v8f;

#define NEG_INF (-__builtin_inff())

// Async global->LDS (CDNA5). Guarded: falls back to load + ds_store staging.
#if defined(__has_builtin)
#if __has_builtin(__builtin_amdgcn_global_load_async_to_lds_b128)
#define USE_ASYNC_LDS 1
#endif
#endif

typedef int v4i __attribute__((vector_size(16)));
typedef __attribute__((address_space(1))) v4i* as1_v4i_p;
typedef __attribute__((address_space(3))) v4i* as3_v4i_p;

__device__ __forceinline__ void stage16(const bf16* gsrc, bf16* ldst) {
#if defined(USE_ASYNC_LDS)
    __builtin_amdgcn_global_load_async_to_lds_b128(
        (as1_v4i_p)gsrc, (as3_v4i_p)ldst, 0, 0);
#else
    *reinterpret_cast<v8bf*>(ldst) = *reinterpret_cast<const v8bf*>(gsrc);
#endif
}

template <int N>
__device__ __forceinline__ void wait_async() {
#if defined(USE_ASYNC_LDS)
#if __has_builtin(__builtin_amdgcn_s_wait_asynccnt)
    __builtin_amdgcn_s_wait_asynccnt(N);
#else
    if (N == 0) asm volatile("s_wait_asynccnt 0x0" ::: "memory");
    else        asm volatile("s_wait_asynccnt 0x1" ::: "memory");
#endif
#endif
}

__device__ __forceinline__ v16bf frag_from2(v8bf a, v8bf b) {
    v16bf r;
#pragma unroll
    for (int i = 0; i < 8; ++i) { r[i] = a[i]; r[i + 8] = b[i]; }
    return r;
}

// A-matrix fragment (16x32, MxK). Lane holds row M = m0+(lane&15).
// elements 0..15 -> K = hi*16 + e.
__device__ __forceinline__ v16bf load_a_frag(const bf16* p, int hi) {
    v8bf lo = *reinterpret_cast<const v8bf*>(p + hi * 16);
    v8bf hh = *reinterpret_cast<const v8bf*>(p + hi * 16 + 8);
    return frag_from2(lo, hh);
}

// B-matrix fragment (32x16, KxN) from Bt stored [N][K] row-major; pointer
// pre-offset by k0 + hi*16: one contiguous 32B run.
__device__ __forceinline__ v16bf load_b_frag(const bf16* p) {
    v8bf lo = *reinterpret_cast<const v8bf*>(p);
    v8bf hh = *reinterpret_cast<const v8bf*>(p + 8);
    return frag_from2(lo, hh);
}

__device__ __forceinline__ v8f wmma_bf16(v16bf a, v16bf b, v8f c) {
    return __builtin_amdgcn_wmma_f32_16x16x32_bf16(
        false, a, false, b, (short)0, c, false, false);
}

// ---------------------------------------------------------------------------
// Weight convert + transpose (coalesced both sides): wt[n*K+k] = bf16(w[k*N+n])
// grid (N/32, K/32), block (32,8)
// ---------------------------------------------------------------------------
__global__ void cvt_t_kernel(const float* __restrict__ w, bf16* __restrict__ wt,
                             int K, int N) {
    __shared__ float tile[32][33];
    const int nb = blockIdx.x * 32, kb = blockIdx.y * 32;
    const int tx = threadIdx.x, ty = threadIdx.y;
#pragma unroll
    for (int i = 0; i < 32; i += 8)
        tile[ty + i][tx] = w[(size_t)(kb + ty + i) * N + nb + tx];
    __syncthreads();
#pragma unroll
    for (int i = 0; i < 32; i += 8)
        wt[(size_t)(nb + ty + i) * K + kb + tx] = (bf16)tile[tx][ty + i];
}

// ---------------------------------------------------------------------------
// Fused LayerNorm over C=1024, up to two (g,b) output pairs (bf16 outputs)
// ---------------------------------------------------------------------------
__global__ void ln_kernel(const float* __restrict__ x,
                          const float* __restrict__ g1, const float* __restrict__ b1,
                          bf16* __restrict__ o1,
                          const float* __restrict__ g2, const float* __restrict__ b2,
                          bf16* __restrict__ o2) {
    const int row  = blockIdx.x;
    const int lane = threadIdx.x & 31;
    const int wav  = threadIdx.x >> 5;
    const float* xr = x + (size_t)row * 1024;

    float s = 0.f, s2 = 0.f;
    for (int i = threadIdx.x; i < 1024; i += 256) {
        float v = xr[i];
        s += v; s2 += v * v;
    }
#pragma unroll
    for (int o = 16; o; o >>= 1) { s += __shfl_xor(s, o); s2 += __shfl_xor(s2, o); }

    __shared__ float shs[8], shs2[8];
    if (lane == 0) { shs[wav] = s; shs2[wav] = s2; }
    __syncthreads();
    if (threadIdx.x == 0) {
        float a = 0.f, b = 0.f;
#pragma unroll
        for (int i = 0; i < 8; ++i) { a += shs[i]; b += shs2[i]; }
        shs[0] = a; shs2[0] = b;
    }
    __syncthreads();
    const float mean = shs[0] * (1.f / 1024.f);
    const float var  = shs2[0] * (1.f / 1024.f) - mean * mean;
    const float rstd = rsqrtf(var + 1e-5f);

    for (int i = threadIdx.x; i < 1024; i += 256) {
        float nv = (xr[i] - mean) * rstd;
        o1[(size_t)row * 1024 + i] = (bf16)(nv * g1[i] + b1[i]);
        if (o2) o2[(size_t)row * 1024 + i] = (bf16)(nv * g2[i] + b2[i]);
    }
}

// ---------------------------------------------------------------------------
// WMMA GEMM with LDS-staged, double-buffered B tile (async global->LDS).
// D[M,N] = act(A[M,K] @ Bt[N,K]^T + bias) (+ epilogue)
// Block: 8 waves, tile 128(M) x 64(N); B tile 64x32 bf16 in LDS, row pitch 40
// elements (80B) for conflict-free ds_load_b128.
// ACT:   0 none, 1 exact GELU, 2 ReLU
// OMODE: 0 bf16 out[row*ldo+col]
//        1 bf16 V-transposed out[((b*16+h)*64+d)*1024 + t]
//        2 f32  out = rscale*resid + val
//        3 f32  out += val
// ---------------------------------------------------------------------------
template <int ACT, int OMODE>
__global__ void gemm_wmma_kernel(const bf16* __restrict__ A, int lda,
                                 const bf16* __restrict__ Bt, int ldb,
                                 const float* __restrict__ bias,
                                 const float* __restrict__ resid, float rscale,
                                 float* __restrict__ outf, bf16* __restrict__ outh,
                                 int ldo, int K) {
    __shared__ __align__(16) bf16 ldsB[2][64][40];

    const int tid  = threadIdx.x;
    const int lane = tid & 31;
    const int wav  = tid >> 5;
    const int hi   = lane >> 4;
    const int l15  = lane & 15;
    const int m0   = blockIdx.y * 128 + wav * 16;
    const int n0   = blockIdx.x * 64;

    // staging: 256 threads cover 64 rows x 4 16B segments
    const int srow = tid >> 2;
    const int sseg = tid & 3;
    const bf16* gstage = Bt + (size_t)(n0 + srow) * ldb + sseg * 8;
    bf16* lstage[2] = { &ldsB[0][srow][sseg * 8], &ldsB[1][srow][sseg * 8] };

    v8f acc[4] = {};
    const bf16* arow = A + (size_t)(m0 + l15) * lda;

    const int steps = K >> 5;
    stage16(gstage, lstage[0]);

    for (int it = 0; it < steps; ++it) {
        const int k0 = it << 5;
        if (it + 1 < steps) {
            stage16(gstage + k0 + 32, lstage[(it + 1) & 1]);
            wait_async<1>();
        } else {
            wait_async<0>();
        }
        __syncthreads();  // staged tile visible to all waves

        v16bf af = load_a_frag(arow + k0, hi);
        const bf16* bb = &ldsB[it & 1][0][hi * 16];
#pragma unroll
        for (int j = 0; j < 4; ++j) {
            v16bf bfr = load_b_frag(bb + (size_t)(j * 16 + l15) * 40);
            acc[j] = wmma_bf16(af, bfr, acc[j]);
        }
        __syncthreads();  // all waves done reading before buffer reuse
    }

#pragma unroll
    for (int j = 0; j < 4; ++j) {
        const int col = n0 + j * 16 + l15;
        const float bb = bias[col];
#pragma unroll
        for (int r = 0; r < 8; ++r) {
            const int row = m0 + r + hi * 8;
            float v = acc[j][r] + bb;
            if (ACT == 1) v = 0.5f * v * (1.f + erff(v * 0.70710678118f));
            else if (ACT == 2) v = fmaxf(v, 0.f);
            const size_t oi = (size_t)row * ldo + col;
            if (OMODE == 0) {
                outh[oi] = (bf16)v;
            } else if (OMODE == 1) {
                const int bi = row >> 10, t = row & 1023;
                const int hh = col >> 6,  d = col & 63;
                outh[(((size_t)(bi * 16 + hh) * 64 + d) << 10) + t] = (bf16)v;
            } else if (OMODE == 2) {
                outf[oi] = rscale * resid[oi] + v;
            } else {
                outf[oi] += v;
            }
        }
    }
}

// ---------------------------------------------------------------------------
// Flash attention (causal), B=4, T=1024, H=16, D=64.
// Q,K: bf16 [B*T, 1024] (b,t,h,d).  Vt: bf16 [b][h][d][t].  Y: bf16 [B*T,1024].
// Block: 8 waves x 16 q-rows = 128 q rows per (b,h). Key blocks of 32.
// ---------------------------------------------------------------------------
__global__ void attn_flash_kernel(const bf16* __restrict__ Q,
                                  const bf16* __restrict__ Km,
                                  const bf16* __restrict__ Vt,
                                  bf16* __restrict__ Y) {
    __shared__ __align__(16) bf16 pbuf[8][16][32];

    const int lane = threadIdx.x & 31;
    const int wav  = threadIdx.x >> 5;
    const int hi   = lane >> 4;
    const int l15  = lane & 15;
    const int b    = blockIdx.z;
    const int h    = blockIdx.y;
    const int q0   = blockIdx.x * 128 + wav * 16;
    const size_t rowbase = (size_t)b * 1024;

    const bf16* qrow = Q + (rowbase + q0 + l15) * 1024 + h * 64;

    v8f acc[4] = {};
    float mrow[8], lrow[8];
#pragma unroll
    for (int r = 0; r < 8; ++r) { mrow[r] = NEG_INF; lrow[r] = 0.f; }

    const int kend = q0 + 16;  // causal: this wave's rows need keys <= q0+15
    for (int kb = 0; kb < kend; kb += 32) {
        // S[16 x 32] = Q(16xD) @ K^T, contraction over d = 64 (2 WMMA K-steps)
        v8f s[2] = {};
#pragma unroll
        for (int d0 = 0; d0 < 64; d0 += 32) {
            v16bf aq = load_a_frag(qrow + d0, hi);
#pragma unroll
            for (int j = 0; j < 2; ++j) {
                const bf16* kk = Km + (rowbase + kb + j * 16 + l15) * 1024
                                    + h * 64 + d0 + hi * 16;
                s[j] = wmma_bf16(aq, load_b_frag(kk), s[j]);
            }
        }

        const float sc = 0.125f;  // 1/sqrt(64)
        const int key0 = kb + l15, key1 = key0 + 16;

        // masked, scaled scores + batched row reductions (8 rows interleaved
        // so the ds_bpermute shuffles issue independently)
        float v0[8], v1[8], mx[8];
#pragma unroll
        for (int r = 0; r < 8; ++r) {
            const int qr = q0 + r + hi * 8;
            v0[r] = (key0 <= qr) ? s[0][r] * sc : NEG_INF;
            v1[r] = (key1 <= qr) ? s[1][r] * sc : NEG_INF;
            mx[r] = fmaxf(v0[r], v1[r]);
        }
#pragma unroll
        for (int o = 8; o; o >>= 1)
#pragma unroll
            for (int r = 0; r < 8; ++r)
                mx[r] = fmaxf(mx[r], __shfl_xor(mx[r], o, 16));

        float corr[8], rs[8];
#pragma unroll
        for (int r = 0; r < 8; ++r) {
            const float mn = fmaxf(mrow[r], mx[r]);
            corr[r] = __expf(mrow[r] - mn);
            mrow[r] = mn;
            const float p0 = __expf(v0[r] - mn);
            const float p1 = __expf(v1[r] - mn);
            rs[r] = p0 + p1;
            pbuf[wav][r + hi * 8][l15]      = (bf16)p0;
            pbuf[wav][r + hi * 8][16 + l15] = (bf16)p1;
        }
#pragma unroll
        for (int o = 8; o; o >>= 1)
#pragma unroll
            for (int r = 0; r < 8; ++r)
                rs[r] += __shfl_xor(rs[r], o, 16);
#pragma unroll
        for (int r = 0; r < 8; ++r) {
            lrow[r] = lrow[r] * corr[r] + rs[r];
#pragma unroll
            for (int j = 0; j < 4; ++j) acc[j][r] *= corr[r];
        }

        // same-wave LDS RAW: make P visible before fragment reload
        asm volatile("s_wait_dscnt 0" ::: "memory");

        // P as A-fragment (16 x 32 over keys), V^T rows as B-fragments
        v16bf pf = load_a_frag(&pbuf[wav][l15][0], hi);
#pragma unroll
        for (int j = 0; j < 4; ++j) {
            const bf16* vv = Vt + (((size_t)(b * 16 + h) * 64 + j * 16 + l15) << 10)
                                + kb + hi * 16;
            acc[j] = wmma_bf16(pf, load_b_frag(vv), acc[j]);
        }
    }

#pragma unroll
    for (int j = 0; j < 4; ++j)
#pragma unroll
        for (int r = 0; r < 8; ++r) {
            const float v = acc[j][r] / lrow[r];
            Y[(rowbase + q0 + r + hi * 8) * 1024 + h * 64 + j * 16 + l15] = (bf16)v;
        }
}

// ---------------------------------------------------------------------------
// Host-side launch
// ---------------------------------------------------------------------------
extern "C" void kernel_launch(void* const* d_in, const int* in_sizes, int n_in,
                              void* d_out, int out_size, void* d_ws, size_t ws_size,
                              hipStream_t stream) {
    (void)in_sizes; (void)n_in; (void)out_size; (void)ws_size;
    const float* x    = (const float*)d_in[0];
    const float* ln1g = (const float*)d_in[1];
    const float* ln1b = (const float*)d_in[2];
    const float* ln2g = (const float*)d_in[3];
    const float* ln2b = (const float*)d_in[4];
    const float* ln3g = (const float*)d_in[5];
    const float* ln3b = (const float*)d_in[6];
    const float* wq = (const float*)d_in[7];   const float* bq = (const float*)d_in[8];
    const float* wk = (const float*)d_in[9];   const float* bk = (const float*)d_in[10];
    const float* wv = (const float*)d_in[11];  const float* bv = (const float*)d_in[12];
    const float* wo = (const float*)d_in[13];  const float* bo = (const float*)d_in[14];
    const float* w1 = (const float*)d_in[15];  const float* b1 = (const float*)d_in[16];
    const float* w2 = (const float*)d_in[17];  const float* b2 = (const float*)d_in[18];
    const float* wd = (const float*)d_in[19];  const float* bd = (const float*)d_in[20];
    const float* wu = (const float*)d_in[21];  const float* bu = (const float*)d_in[22];
    float* out = (float*)d_out;

    const int M = 4096;   // B*T
    const int C = 1024;
    const int F = 4096;   // 4C
    const int Ad = 64;    // adapter rank

    char* wsp = (char*)d_ws;
    auto alloc = [&](size_t bytes) -> char* {
        char* p = wsp;
        wsp += (bytes + 255) & ~(size_t)255;
        return p;
    };
    bf16* h_bf  = (bf16*)alloc((size_t)M * C * 2);   // ln1(x)
    bf16* n3_bf = (bf16*)alloc((size_t)M * C * 2);   // ln3(x)
    bf16* m_bf  = (bf16*)alloc((size_t)M * C * 2);   // ln2(hidden)
    bf16* wq_t  = (bf16*)alloc((size_t)C * C * 2);
    bf16* wk_t  = (bf16*)alloc((size_t)C * C * 2);
    bf16* wv_t  = (bf16*)alloc((size_t)C * C * 2);
    bf16* wo_t  = (bf16*)alloc((size_t)C * C * 2);
    bf16* w1_t  = (bf16*)alloc((size_t)C * F * 2);
    bf16* w2_t  = (bf16*)alloc((size_t)C * F * 2);
    bf16* wd_t  = (bf16*)alloc((size_t)C * Ad * 2);
    bf16* wu_t  = (bf16*)alloc((size_t)C * Ad * 2);
    bf16* qbuf  = (bf16*)alloc((size_t)M * C * 2);
    bf16* kbuf  = (bf16*)alloc((size_t)M * C * 2);
    bf16* vtbuf = (bf16*)alloc((size_t)M * C * 2);   // [b][h][d][t]
    bf16* ybuf  = (bf16*)alloc((size_t)M * C * 2);
    float* hidden = (float*)alloc((size_t)M * C * 4);
    bf16* g1buf = (bf16*)alloc((size_t)M * F * 2);
    bf16* adbuf = (bf16*)alloc((size_t)M * Ad * 2);

    // 1) weight convert + transpose to bf16 [N][K] (tiled, coalesced)
    auto cvt = [&](const float* w, bf16* wt, int K, int N) {
        cvt_t_kernel<<<dim3(N / 32, K / 32), dim3(32, 8), 0, stream>>>(w, wt, K, N);
    };
    cvt(wq, wq_t, C, C);  cvt(wk, wk_t, C, C);
    cvt(wv, wv_t, C, C);  cvt(wo, wo_t, C, C);
    cvt(w1, w1_t, C, F);  cvt(w2, w2_t, F, C);
    cvt(wd, wd_t, C, Ad); cvt(wu, wu_t, Ad, C);

    // 2) ln1(x) and ln3(x) fused
    ln_kernel<<<M, 256, 0, stream>>>(x, ln1g, ln1b, h_bf, ln3g, ln3b, n3_bf);

    // 3) Q/K/V projections
    dim3 gCC(C / 64, M / 128);
    gemm_wmma_kernel<0, 0><<<gCC, 256, 0, stream>>>(h_bf, C, wq_t, C, bq,
                                                    nullptr, 0.f, nullptr, qbuf, C, C);
    gemm_wmma_kernel<0, 0><<<gCC, 256, 0, stream>>>(h_bf, C, wk_t, C, bk,
                                                    nullptr, 0.f, nullptr, kbuf, C, C);
    gemm_wmma_kernel<0, 1><<<gCC, 256, 0, stream>>>(h_bf, C, wv_t, C, bv,
                                                    nullptr, 0.f, nullptr, vtbuf, C, C);

    // 4) causal flash attention
    attn_flash_kernel<<<dim3(8, 16, 4), 256, 0, stream>>>(qbuf, kbuf, vtbuf, ybuf);

    // 5) output projection + residual -> hidden (f32)
    gemm_wmma_kernel<0, 2><<<gCC, 256, 0, stream>>>(ybuf, C, wo_t, C, bo,
                                                    x, 1.f, hidden, nullptr, C, C);

    // 6) ln2(hidden)
    ln_kernel<<<M, 256, 0, stream>>>(hidden, ln2g, ln2b, m_bf,
                                     nullptr, nullptr, nullptr);

    // 7) MLP up + exact GELU (bf16)
    dim3 gCF(F / 64, M / 128);
    gemm_wmma_kernel<1, 0><<<gCF, 256, 0, stream>>>(m_bf, C, w1_t, C, b1,
                                                    nullptr, 0.f, nullptr, g1buf, F, C);

    // 8) MLP down; out = 2*hidden + mlp_out  (= hidden + forward_states)
    gemm_wmma_kernel<0, 2><<<gCC, 256, 0, stream>>>(g1buf, F, w2_t, F, b2,
                                                    hidden, 2.f, out, nullptr, C, F);

    // 9) adapter down + ReLU (N=64)
    dim3 gA(Ad / 64, M / 128);
    gemm_wmma_kernel<2, 0><<<gA, 256, 0, stream>>>(n3_bf, C, wd_t, C, bd,
                                                   nullptr, 0.f, nullptr, adbuf, Ad, C);

    // 10) adapter up, accumulate into out (K=64)
    gemm_wmma_kernel<0, 3><<<gCC, 256, 0, stream>>>(adbuf, Ad, wu_t, Ad, bu,
                                                    nullptr, 0.f, out, nullptr, C, Ad);
}
